// DK_MSA_10024453669013
// MI455X (gfx1250) — matrix operand order repaired and checked by reading, not compiled, not run on hardware
//
#include <hip/hip_runtime.h>
#include <hip/hip_bf16.h>

// MI455X (gfx1250) implementation.
#define BB  16      // batch
#define TT  16      // time
#define NN  184     // spatial
#define DD  128     // model dim
#define HH  8       // heads
#define HD  16      // head dim
#define NP  192     // N padded to 12*16
#define KP  32      // K padding for 16-wide K dims
#define ROWS (BB*TT*NN)          // 47104
#define ROW_TILES (ROWS/16)      // 2944
#define QKV_ELEMS (HH*BB*TT*NN*HD)   // 6,029,312 elems per tensor

typedef __attribute__((ext_vector_type(16))) _Float16 v16h;
typedef __attribute__((ext_vector_type(8)))  float    v8f;

union H16 { v16h v; _Float16 e[16]; };
union F8  { v8f  v; float    e[8];  };

// ---- WMMA fragment index maps (CDNA5 ISA 7.12.2, 16-bit operands, wave32) ----
__device__ __forceinline__ int kA_of(int lane, int i) {
  int v = i >> 1, p = i & 1;
  int base = (v < 4) ? (v << 1) : (16 + ((v - 4) << 1));
  return base + ((lane >= 16) ? 8 : 0) + p;
}
__device__ __forceinline__ int kB_of(int lane, int i) {
  return ((i < 8) ? i : (8 + i)) + ((lane >= 16) ? 8 : 0);
}

__device__ __forceinline__ v8f wmma_f16(v16h a, v16h b, v8f c) {
  return __builtin_amdgcn_wmma_f32_16x16x32_f16(false, a, false, b, (short)0, c,
                                                false, false);
}

// =====================================================================
// 16-row x 128-col GEMM  out = scale*(X @ W^T + bias)
//   X_F16:     input rows already f16 (16B-vector staged), else f32+convert
//   OUT_SPLIT: 1 -> f16 scatter into [H*B][T][N][HD]; 0 -> f32 row-major
// scale folded into the W/bias f16 conversion (free; used to pre-apply the
// attention 1/sqrt(HD) to q so the attention hot loop has no scaling).
// =====================================================================
template<bool X_F16, int OUT_SPLIT>
__global__ void gemm_d128(const void* __restrict__ Xv, const float* __restrict__ W,
                          const float* __restrict__ bias, void* __restrict__ outv,
                          float scale) {
  __shared__ _Float16 Xs[16 * 128];
  __shared__ _Float16 Ws[128 * 128];
  const int tid  = threadIdx.x;
  const int lane = tid & 31;
  const int w    = tid >> 5;              // 0..7
  const unsigned R0 = blockIdx.x * 16u;

  if constexpr (X_F16) {
    const uint4* src = (const uint4*)((const _Float16*)Xv + (size_t)R0 * 128u);
    ((uint4*)Xs)[tid] = src[tid];
  } else {
    const float* X = (const float*)Xv;
    for (int idx = tid; idx < 16 * 128; idx += 256)
      Xs[idx] = (_Float16)X[R0 * 128u + (unsigned)idx];
  }
  for (int idx = tid; idx < 128 * 128; idx += 256)
    Ws[idx] = (_Float16)(W[idx] * scale);
  __syncthreads();

  const int m16 = lane & 15;
  const int nc  = w * 16 + m16;
  F8 c;
  {
    float bv = bias[nc] * scale;
#pragma unroll
    for (int r = 0; r < 8; ++r) c.e[r] = bv;
  }
#pragma unroll
  for (int kb = 0; kb < 128; kb += 32) {
    H16 a, b;
#pragma unroll
    for (int i = 0; i < 16; ++i) {
      a.e[i] = Xs[m16 * 128 + kb + kA_of(lane, i)];
      b.e[i] = Ws[nc * 128 + kb + kB_of(lane, i)];   // B[k][n] = W[n][k]
    }
    c.v = wmma_f16(a.v, b.v, c.v);
  }

  const int hi = (lane >= 16) ? 8 : 0;
  if constexpr (OUT_SPLIT) {
    _Float16* out = (_Float16*)outv;
    const unsigned bt0 = R0 / (unsigned)NN;        // 32-bit magic-mul udiv, once
    const unsigned nn0 = R0 - bt0 * (unsigned)NN;
    const int h = nc >> 4, hd = nc & 15;
#pragma unroll
    for (int r = 0; r < 8; ++r) {
      unsigned nn = nn0 + (unsigned)(r + hi);
      unsigned bt = bt0;
      if (nn >= (unsigned)NN) { nn -= (unsigned)NN; bt += 1u; }  // cndmask
      unsigned t = bt & 15u, b = bt >> 4;          // TT=BB=16
      unsigned idx = ((((unsigned)h * BB + b) * TT + t) * NN + nn) * HD + (unsigned)hd;
      out[idx] = (_Float16)c.e[r];
    }
  } else {
    float* out = (float*)outv;
#pragma unroll
    for (int r = 0; r < 8; ++r)
      out[(R0 + (unsigned)(r + hi)) * DD + (unsigned)nc] = c.e[r];
  }
}

// =====================================================================
// adp = ne1@ne2 [184x184]; row-softmax -> f16 adp_row;
// col-softmax transposed -> f16 adp_colT. Single workgroup, L2-resident.
// =====================================================================
__global__ void adp_kernel(const float* __restrict__ ne1, const float* __restrict__ ne2,
                           float* __restrict__ adp_buf, _Float16* __restrict__ adp_row,
                           _Float16* __restrict__ adp_colT) {
  const int tid = threadIdx.x;
  for (unsigned idx = tid; idx < NN * NN; idx += 256) {
    unsigned i = idx / NN, j = idx - i * NN;
    float acc = 0.f;
#pragma unroll
    for (int e = 0; e < 10; ++e) acc += ne1[i * 10 + e] * ne2[e * NN + j];
    adp_buf[idx] = acc;
  }
  __syncthreads();
  if (tid < NN) {
    float m = -3.4e38f;
    for (int j = 0; j < NN; ++j) m = fmaxf(m, adp_buf[tid * NN + j]);
    float s = 0.f;
    for (int j = 0; j < NN; ++j) s += __expf(adp_buf[tid * NN + j] - m);
    float inv = 1.f / s;
    for (int j = 0; j < NN; ++j)
      adp_row[tid * NN + j] = (_Float16)(__expf(adp_buf[tid * NN + j] - m) * inv);
    float cm = -3.4e38f;
    for (int k = 0; k < NN; ++k) cm = fmaxf(cm, adp_buf[k * NN + tid]);
    float cs = 0.f;
    for (int k = 0; k < NN; ++k) cs += __expf(adp_buf[k * NN + tid] - cm);
    float cinv = 1.f / cs;
    for (int k = 0; k < NN; ++k)
      adp_colT[tid * NN + k] = (_Float16)(__expf(adp_buf[k * NN + tid] - cm) * cinv);
  }
}

// =====================================================================
// Fused attention per (hb = h*B+b, t). 12 waves, ~229 KB LDS.
// All fragment loads are UNCONDITIONAL (zero-padded layouts) -> vectorized
// ds_load_b128. q arrives pre-scaled by 1/sqrt(HD).
// =====================================================================
__global__ void attn_kernel(const _Float16* __restrict__ qh, const _Float16* __restrict__ kh,
                            const _Float16* __restrict__ vh,
                            const _Float16* __restrict__ adp_row,
                            const _Float16* __restrict__ adp_colT,
                            const float* __restrict__ Wm, const float* __restrict__ bm,
                            _Float16* __restrict__ merged) {
  extern __shared__ char smem[];
  _Float16* qs   = (_Float16*)smem;                // [NP][KP] K-padded
  _Float16* ksm  = qs  + NP * KP;                  // [NP][KP]
  _Float16* vsm  = ksm + NP * KP;                  // [NP][HD] row-padded
  _Float16* Wms  = vsm + NP * HD;                  // [4][KP][16] K-padded
  float* S       = (float*)(smem + (2 * NP * KP + NP * HD + 4 * KP * 16) * 2);
  float* rmax    = S + NP * NP;                    // [NP] (padded entries safe)
  float* rsinv   = rmax  + NP;
  float* cmax    = rsinv + NP;
  float* csinv   = cmax  + NP;
  _Float16* oscr = (_Float16*)(csinv + NP);        // 12 waves * 4 * [16][KP]

  const int tid  = threadIdx.x;
  const int lane = tid & 31;
  const int w    = tid >> 5;                       // 0..11 (row tile)
  const int m16  = lane & 15;
  const int hi   = (lane >= 16) ? 8 : 0;

  const unsigned blk = blockIdx.x;                 // hb*T + t ; T=16, B=16
  const unsigned t   = blk & 15u;
  const unsigned hb  = blk >> 4;
  const unsigned h   = hb >> 4, b = hb & 15u;
  const size_t base  = (size_t)blk * (NN * HD);    // halves

  // ---- staging: 16B chunks; zero-fill K pad and row pad
  {
    typedef uint4 C16;
    const C16 z = {0u, 0u, 0u, 0u};
    C16 qv = z, kv = z, vv = z;
    if (tid < (NN * HD) / 8) {                     // 368 data chunks
      qv = ((const C16*)(qh + base))[tid];
      kv = ((const C16*)(kh + base))[tid];
      vv = ((const C16*)(vh + base))[tid];
    }
    const int row = tid >> 1, c8 = (tid & 1) * 8;  // row 0..191
    *(C16*)(qs  + row * KP + c8)      = qv;        // data (or zeros) low half
    *(C16*)(qs  + row * KP + 16 + c8) = z;         // K-pad high half
    *(C16*)(ksm + row * KP + c8)      = kv;
    *(C16*)(ksm + row * KP + 16 + c8) = z;
    *(C16*)(vsm + tid * 8)            = vv;        // [192][16], rows>=184 zero
  }
  // Wm -> LDS, K-padded: Wms[p][k][n] = (k<16) ? Wm[n*64 + p*16 + k] : 0
  for (int idx = tid; idx < 4 * KP * 16; idx += 384) {
    int p = idx >> 9, rem = idx & 511, k = rem >> 4, n = rem & 15;
    Wms[idx] = (k < HD) ? (_Float16)Wm[n * 64 + p * HD + k] : (_Float16)0.f;
  }
  __syncthreads();

  // ---- Phase 1: S = q' k^T (q pre-scaled; full 192x192 incl. zero padding)
  for (int tj = 0; tj < 12; ++tj) {
    H16 a, bfr;
#pragma unroll
    for (int i = 0; i < 16; ++i) {
      a.e[i]   = qs [(w  * 16 + m16) * KP + kA_of(lane, i)];
      bfr.e[i] = ksm[(tj * 16 + m16) * KP + kB_of(lane, i)];
    }
    F8 c;
#pragma unroll
    for (int r = 0; r < 8; ++r) c.e[r] = 0.f;
    c.v = wmma_f16(a.v, bfr.v, c.v);
#pragma unroll
    for (int r = 0; r < 8; ++r)
      S[(w * 16 + r + hi) * NP + tj * 16 + m16] = c.e[r];
  }
  __syncthreads();

  // ---- Phase 2: softmax stats; padded rows get safe (0,0) stats
  if (tid < NN) {
    float m = -3.4e38f;
    for (int j = 0; j < NN; ++j) m = fmaxf(m, S[tid * NP + j]);
    float s = 0.f;
    for (int j = 0; j < NN; ++j) s += __expf(S[tid * NP + j] - m);
    rmax[tid] = m; rsinv[tid] = 1.f / s;
    float cm = -3.4e38f;
    for (int i = 0; i < NN; ++i) cm = fmaxf(cm, S[i * NP + tid]);
    float cs = 0.f;
    for (int i = 0; i < NN; ++i) cs += __expf(S[i * NP + tid] - cm);
    cmax[tid] = cm; csinv[tid] = 1.f / cs;
  } else if (tid < NP) {
    rmax[tid] = 0.f; rsinv[tid] = 0.f; cmax[tid] = 0.f; csinv[tid] = 0.f;
  }
  __syncthreads();

  // ---- Phase 3: o1..o4 for row-tile w; 6 K-steps; all loads unconditional.
  // Padded K rows of vsm are zero, so junk P/adp values there contribute 0.
  F8 a1, a2, a3, a4;
#pragma unroll
  for (int r = 0; r < 8; ++r) { a1.e[r] = 0.f; a2.e[r] = 0.f; a3.e[r] = 0.f; a4.e[r] = 0.f; }
  const int iRow = w * 16 + m16;                   // < 192
  const int iR   = (iRow < NN) ? iRow : (NN - 1);  // clamp for adp reads
  const float rm = rmax[iRow], ri = rsinv[iRow];
  const float cm = cmax[iRow], ci = csinv[iRow];
  for (int kb0 = 0; kb0 < NP; kb0 += 32) {
    H16 bv;
#pragma unroll
    for (int i = 0; i < 16; ++i)
      bv.e[i] = vsm[(kb0 + kB_of(lane, i)) * HD + m16];
    H16 f1, f2, f3, f4;
#pragma unroll
    for (int i = 0; i < 16; ++i) {
      int j  = kb0 + kA_of(lane, i);
      int jc = (j < NN) ? j : (NN - 1);            // clamp for adp reads
      f1.e[i] = (_Float16)(__expf(S[iRow * NP + j] - rm) * ri);  // P_row[i][j]
      f2.e[i] = (_Float16)(__expf(S[j * NP + iRow] - cm) * ci);  // P_col^T[i][j]
      f3.e[i] = adp_row [iR * NN + jc];
      f4.e[i] = adp_colT[iR * NN + jc];
    }
    a1.v = wmma_f16(f1.v, bv.v, a1.v);
    a2.v = wmma_f16(f2.v, bv.v, a2.v);
    a3.v = wmma_f16(f3.v, bv.v, a3.v);
    a4.v = wmma_f16(f4.v, bv.v, a4.v);
  }

  // ---- Phase 4: concat @ Wm^T + bm folded as 4 accumulated WMMAs.
  // o tiles staged to wave-private K-padded f16 scratch (zeros in pad).
  _Float16* myscr = oscr + w * (4 * 16 * KP);
#pragma unroll
  for (int r = 0; r < 8; ++r) {
    int m = r + hi;
    myscr[0 * 512 + m * KP + m16]      = (_Float16)a1.e[r];
    myscr[0 * 512 + m * KP + 16 + m16] = (_Float16)0.f;
    myscr[1 * 512 + m * KP + m16]      = (_Float16)a2.e[r];
    myscr[1 * 512 + m * KP + 16 + m16] = (_Float16)0.f;
    myscr[2 * 512 + m * KP + m16]      = (_Float16)a3.e[r];
    myscr[2 * 512 + m * KP + 16 + m16] = (_Float16)0.f;
    myscr[3 * 512 + m * KP + m16]      = (_Float16)a4.e[r];
    myscr[3 * 512 + m * KP + 16 + m16] = (_Float16)0.f;
  }
  F8 co;
  {
    float bmv = bm[m16];
#pragma unroll
    for (int r = 0; r < 8; ++r) co.e[r] = bmv;
  }
#pragma unroll
  for (int p = 0; p < 4; ++p) {
    H16 af, bf;
#pragma unroll
    for (int i = 0; i < 16; ++i) {
      af.e[i] = myscr[p * 512 + m16 * KP + kA_of(lane, i)];
      bf.e[i] = Wms[(p * KP + kB_of(lane, i)) * 16 + m16];
    }
    co.v = wmma_f16(af.v, bf.v, co.v);
  }
  // head-merged f16 store: merged[b][t][n][h*16+hd]
  const unsigned mbase = ((b * TT + t) * (unsigned)NN) * DD + h * HD + (unsigned)m16;
#pragma unroll
  for (int r = 0; r < 8; ++r) {
    int n = w * 16 + r + hi;
    if (n < NN) merged[mbase + (unsigned)n * DD] = (_Float16)co.e[r];
  }
}

extern "C" void kernel_launch(void* const* d_in, const int* in_sizes, int n_in,
                              void* d_out, int out_size, void* d_ws, size_t ws_size,
                              hipStream_t stream) {
  const float* query = (const float*)d_in[0];
  const float* key   = (const float*)d_in[1];
  const float* value = (const float*)d_in[2];
  const float* Wq = (const float*)d_in[3];  const float* bq = (const float*)d_in[4];
  const float* Wk = (const float*)d_in[5];  const float* bk = (const float*)d_in[6];
  const float* Wv = (const float*)d_in[7];  const float* bv = (const float*)d_in[8];
  const float* Wm = (const float*)d_in[9];  const float* bm = (const float*)d_in[10];
  const float* Wo = (const float*)d_in[11]; const float* bo = (const float*)d_in[12];
  const float* ne1 = (const float*)d_in[13];
  const float* ne2 = (const float*)d_in[14];
  // d_in[15] = dim (harness uses dim==2; this implements that path)

  // workspace: f16 intermediates (halves HBM round-trip traffic)
  char* wsb = (char*)d_ws;
  const size_t qkv_bytes = (size_t)QKV_ELEMS * sizeof(_Float16);  // 16B-aligned
  _Float16* qh     = (_Float16*)(wsb);
  _Float16* kh     = (_Float16*)(wsb + 1 * qkv_bytes);
  _Float16* vh     = (_Float16*)(wsb + 2 * qkv_bytes);
  _Float16* merged = (_Float16*)(wsb + 3 * qkv_bytes);
  float*    adp_buf  = (float*)(wsb + 4 * qkv_bytes);
  _Float16* adp_row  = (_Float16*)(adp_buf + NN * NN);
  _Float16* adp_colT = adp_row + NN * NN;

  adp_kernel<<<1, 256, 0, stream>>>(ne1, ne2, adp_buf, adp_row, adp_colT);

  // q projection pre-applies attention scale 1/sqrt(HD) = 0.25
  gemm_d128<false, 1><<<ROW_TILES, 256, 0, stream>>>(query, Wq, bq, qh, 0.25f);
  gemm_d128<false, 1><<<ROW_TILES, 256, 0, stream>>>(key,   Wk, bk, kh, 1.0f);
  gemm_d128<false, 1><<<ROW_TILES, 256, 0, stream>>>(value, Wv, bv, vh, 1.0f);

  const int smem_bytes = (2 * NP * KP + NP * HD + 4 * KP * 16) * 2  // q,k,v,Wm f16
                       + NP * NP * 4                                // S f32
                       + 4 * NP * 4                                 // stats
                       + 12 * 4 * 16 * KP * 2;                      // o scratch f16
  (void)hipFuncSetAttribute((const void*)attn_kernel,
                            hipFuncAttributeMaxDynamicSharedMemorySize, smem_bytes);
  attn_kernel<<<HH * BB * TT, 384, smem_bytes, stream>>>(
      qh, kh, vh, adp_row, adp_colT, Wm, bm, merged);

  gemm_d128<true, 0><<<ROW_TILES, 256, 0, stream>>>(merged, Wo, bo, (float*)d_out, 1.0f);
}
